// ConvEnhanced_65481071406538
// MI455X (gfx1250) — compile-verified, matrix operand order checked by reading, not run on hardware
//
#include <hip/hip_runtime.h>
#include <hip/hip_bf16.h>

// ---------------------------------------------------------------------------
// ConvEnhanced: 0.5*mean(sigmoid(conv3x3(x,W)+b)) + 0.5*mean(win3x3(sin^2(pi x/2)))/9
//   x: [32,3,384,384] f32, W: [64,3,3,3] f32, b: [64] f32 -> scalar f32
//
// Classical path: implicit GEMM on v_wmma_f32_16x16x32_f16, K=27 padded to 32.
// K-slot permutation chosen so each lane-half owns memory-contiguous runs:
//   lanes 0-15  <- patch reals 0..15   (rows of ch0 + start of ch1)
//   lanes 16-31 <- patch reals 16..26  (+5 zero pad slots)
// => A-fragment built from b96/b64 loads at immediate offsets, no masks.
// Sigmoid folded to one fma + v_tanh_f32 + add (affine part deferred).
// Quantum path collapsed to a weighted elementwise reduction (cos identity).
// ---------------------------------------------------------------------------

typedef __attribute__((ext_vector_type(16))) _Float16 v16h;
typedef __attribute__((ext_vector_type(8)))  float    v8f;

#define BATCH 32
#define CIN   3
#define COUT  64
#define HH    384
#define WW    384
#define HO    382
#define WO    382
#define KRED  27                 // CIN*3*3
#define HWSZ  (HH * WW)          // 147456

struct __align__(4) f3v { float a, b, c; };
struct __align__(4) f2v { float a, b; };

__device__ __forceinline__ float wave_reduce_add(float v) {
    #pragma unroll
    for (int off = 16; off > 0; off >>= 1)
        v += __shfl_xor(v, off, 32);
    return v;
}

__device__ __forceinline__ float hw_tanh(float v) {
#if __has_builtin(__builtin_amdgcn_tanhf)
    return __builtin_amdgcn_tanhf(v);
#else
    float t;
    asm volatile("v_tanh_f32 %0, %1\n\tv_nop" : "=v"(t) : "v"(v));
    return t;
#endif
}

__global__ void ce_init_kernel(double* acc) {
    acc[0] = 0.0;
    acc[1] = 0.0;
}

// --------------------------- classical path (WMMA) -------------------------
__global__ __launch_bounds__(256) void ce_conv_sigmoid_kernel(
    const float* __restrict__ x,
    const float* __restrict__ w,
    const float* __restrict__ bias,
    double* __restrict__ acc)
{
    const int lane  = threadIdx.x & 31;
    const int wave  = threadIdx.x >> 5;
    const int wpb   = blockDim.x >> 5;
    const int wid   = blockIdx.x * wpb + wave;
    const int nwave = gridDim.x * wpb;

    const int  lrow  = lane & 15;        // A row / B col / D col for this lane
    const bool upper = (lane >> 4) & 1;  // K-half ownership

    // B fragments (weights), 4 N-tiles of 16 out-channels. Same K permutation
    // as A: lower half slots = reals 0..15, upper half slots = reals 16..26.
    v16h bfrag[4];
    #pragma unroll
    for (int nt = 0; nt < 4; ++nt) {
        const float* wp = w + (nt * 16 + lrow) * KRED;
        v16h bf;
        if (!upper) {
            #pragma unroll
            for (int e = 0; e < 16; ++e) bf[e] = (_Float16)wp[e];
        } else {
            #pragma unroll
            for (int e = 0; e < 16; ++e)
                bf[e] = (e < 11) ? (_Float16)wp[16 + e] : (_Float16)0.0f;
        }
        bfrag[nt] = bf;
    }

    // Pre-halved bias for this lane's D column: tanh(0.5*cf + 0.5*b).
    float bc2[4];
    #pragma unroll
    for (int nt = 0; nt < 4; ++nt) bc2[nt] = 0.5f * bias[nt * 16 + lrow];

    const int nTiles = (BATCH * HO * WO) / 16;   // 291848 exactly
    float tsum  = 0.0f;   // sum of tanh values
    int   tdone = 0;      // tiles processed (for deferred affine part)

    for (int t = wid; t < nTiles; t += nwave) {
        int m  = t * 16 + lrow;
        int b  = m / (HO * WO);
        int r  = m % (HO * WO);
        int oh = r / WO;
        int ow = r % WO;
        int base = b * (CIN * HWSZ) + oh * WW + ow;
        const float* xb = x + base;

        // Speculative prefetch of (approximately) the next grid-stride tile.
        if (t + nwave < nTiles)
            __builtin_prefetch(xb + nwave * 16, 0, 3);

        v16h af;
        if (!upper) {
            // reals 0..15: ch0 rows 0..2, ch1 rows 0..1, first of ch1 row 2
            f3v r0 = *(const f3v*)(xb);
            f3v r1 = *(const f3v*)(xb + WW);
            f3v r2 = *(const f3v*)(xb + 2 * WW);
            f3v r3 = *(const f3v*)(xb + HWSZ);
            f3v r4 = *(const f3v*)(xb + HWSZ + WW);
            float r5 = xb[HWSZ + 2 * WW];
            af[0]  = (_Float16)r0.a; af[1]  = (_Float16)r0.b; af[2]  = (_Float16)r0.c;
            af[3]  = (_Float16)r1.a; af[4]  = (_Float16)r1.b; af[5]  = (_Float16)r1.c;
            af[6]  = (_Float16)r2.a; af[7]  = (_Float16)r2.b;
            af[8]  = (_Float16)r2.c;
            af[9]  = (_Float16)r3.a; af[10] = (_Float16)r3.b; af[11] = (_Float16)r3.c;
            af[12] = (_Float16)r4.a; af[13] = (_Float16)r4.b; af[14] = (_Float16)r4.c;
            af[15] = (_Float16)r5;
        } else {
            // reals 16..26: rest of ch1 row 2, ch2 rows 0..2; pads -> 0
            f2v r5 = *(const f2v*)(xb + HWSZ + 2 * WW + 1);
            f3v r6 = *(const f3v*)(xb + 2 * HWSZ);
            f3v r7 = *(const f3v*)(xb + 2 * HWSZ + WW);
            f3v r8 = *(const f3v*)(xb + 2 * HWSZ + 2 * WW);
            af[0]  = (_Float16)r5.a; af[1]  = (_Float16)r5.b;
            af[2]  = (_Float16)r6.a; af[3]  = (_Float16)r6.b; af[4]  = (_Float16)r6.c;
            af[5]  = (_Float16)r7.a; af[6]  = (_Float16)r7.b; af[7]  = (_Float16)r7.c;
            af[8]  = (_Float16)r8.a; af[9]  = (_Float16)r8.b; af[10] = (_Float16)r8.c;
            af[11] = (_Float16)0.0f; af[12] = (_Float16)0.0f; af[13] = (_Float16)0.0f;
            af[14] = (_Float16)0.0f; af[15] = (_Float16)0.0f;
        }

        #pragma unroll
        for (int nt = 0; nt < 4; ++nt) {
            v8f cf = {};
            cf = __builtin_amdgcn_wmma_f32_16x16x32_f16(
                    false, af, false, bfrag[nt], (short)0, cf, false, false);
            #pragma unroll
            for (int j = 0; j < 8; ++j)
                tsum += hw_tanh(fmaf(0.5f, cf[j], bc2[nt]));   // fma + tanh + add
        }
        ++tdone;
    }

    // sigmoid sum = 0.5*count + 0.5*sum(tanh); count = 32 elements per tile.
    float lsum = fmaf(16.0f, (float)tdone, 0.5f * tsum);

    // Block reduction -> one f64 atomic per block.
    __shared__ float ssum[8];
    float ws = wave_reduce_add(lsum);
    if (lane == 0) ssum[wave] = ws;
    __syncthreads();
    if (threadIdx.x == 0) {
        float t = 0.0f;
        for (int i = 0; i < wpb; ++i) t += ssum[i];
        unsafeAtomicAdd(&acc[0], (double)t);
    }
}

// --------------------------- quantum path ---------------------------------
// mean(win3x3(sin^2(pi x/2)))/9 == sum_e sin^2(pi x_e/2)*cntH(h)*cntW(w)
//                                   / (B*C*Ho*Wo*9),  cnt(i)=min(i+1,3,384-i)
// sin^2(pi x/2) = 0.5 - 0.5*cos(pi x)  -> one v_cos_f32 + FMA per element.
__device__ __forceinline__ float edge_cnt(int i) {
    int c = i + 1;
    if (c > 3) c = 3;
    int d = HH - i;
    if (d < c) c = d;
    return (float)c;
}

__global__ __launch_bounds__(256) void ce_quantum_kernel(
    const float* __restrict__ x, double* __restrict__ acc)
{
    const int n4     = (BATCH * CIN * HH * WW) / 4;  // WW % 4 == 0
    const int tid    = blockIdx.x * blockDim.x + threadIdx.x;
    const int stride = gridDim.x * blockDim.x;
    const float4* x4 = (const float4*)x;

    float lsum = 0.0f;
    for (int i = tid; i < n4; i += stride) {
        int   row = i / (WW / 4);            // global row over B*C*H
        int   w0  = (i - row * (WW / 4)) * 4;
        int   h   = row % HH;
        float wh  = edge_cnt(h);
        float4 v  = x4[i];
        float s0 = fmaf(-0.5f, __cosf(3.14159265359f * v.x), 0.5f);
        float s1 = fmaf(-0.5f, __cosf(3.14159265359f * v.y), 0.5f);
        float s2 = fmaf(-0.5f, __cosf(3.14159265359f * v.z), 0.5f);
        float s3 = fmaf(-0.5f, __cosf(3.14159265359f * v.w), 0.5f);
        lsum += wh * (s0 * edge_cnt(w0)     + s1 * edge_cnt(w0 + 1) +
                      s2 * edge_cnt(w0 + 2) + s3 * edge_cnt(w0 + 3));
    }

    __shared__ float ssum[8];
    const int lane = threadIdx.x & 31;
    const int wave = threadIdx.x >> 5;
    float ws = wave_reduce_add(lsum);
    if (lane == 0) ssum[wave] = ws;
    __syncthreads();
    if (threadIdx.x == 0) {
        float t = 0.0f;
        const int wpb = blockDim.x >> 5;
        for (int i = 0; i < wpb; ++i) t += ssum[i];
        unsafeAtomicAdd(&acc[1], (double)t);
    }
}

// --------------------------- combine ---------------------------------------
__global__ void ce_combine_kernel(const double* __restrict__ acc,
                                  float* __restrict__ out)
{
    // classical denom: 32*64*382*382 ; quantum denom: 32*3*382*382*9
    double cm = acc[0] / 298852352.0;
    double qm = acc[1] / 126078336.0;
    out[0] = (float)(0.5 * cm + 0.5 * qm);
}

// ---------------------------------------------------------------------------
extern "C" void kernel_launch(void* const* d_in, const int* in_sizes, int n_in,
                              void* d_out, int out_size, void* d_ws, size_t ws_size,
                              hipStream_t stream) {
    const float* x = (const float*)d_in[0];
    const float* W = (const float*)d_in[1];
    const float* b = (const float*)d_in[2];
    float* out  = (float*)d_out;
    double* acc = (double*)d_ws;

    ce_init_kernel<<<1, 1, 0, stream>>>(acc);
    ce_conv_sigmoid_kernel<<<2048, 256, 0, stream>>>(x, W, b, acc);
    ce_quantum_kernel<<<2048, 256, 0, stream>>>(x, acc);
    ce_combine_kernel<<<1, 1, 0, stream>>>(acc, out);
}
